// VectorWeightedAttention_23905787969985
// MI455X (gfx1250) — matrix operand order repaired and checked by reading, not compiled, not run on hardware
//
#include <hip/hip_runtime.h>
#include <cmath>
#include <cstdint>

typedef __bf16 bf16;
typedef __attribute__((ext_vector_type(16))) __bf16 v16bf;
typedef __attribute__((ext_vector_type(8)))  __bf16 v8bf;
typedef __attribute__((ext_vector_type(8)))  float  v8f;

#define HIDDEN 512
#define BATCH  64

__device__ inline v16bf join16(v8bf lo, v8bf hi) {
    v16bf r;
    #pragma unroll
    for (int e = 0; e < 8; ++e) { r[e] = lo[e]; r[8 + e] = hi[e]; }
    return r;
}

// ---------------------------------------------------------------------------
// Kernel 0a: elementwise f32 -> bf16 (Q and K), 8 elements/thread
// ---------------------------------------------------------------------------
__global__ __launch_bounds__(256)
void cvt_bf16_kernel(const float* __restrict__ in, bf16* __restrict__ out, size_t n)
{
    size_t i = ((size_t)blockIdx.x * blockDim.x + threadIdx.x) * 8;
    if (i >= n) return;
    const float4* p = (const float4*)(in + i);
    float4 a = p[0], b = p[1];
    v8bf r;
    r[0] = (bf16)a.x; r[1] = (bf16)a.y; r[2] = (bf16)a.z; r[3] = (bf16)a.w;
    r[4] = (bf16)b.x; r[5] = (bf16)b.y; r[6] = (bf16)b.z; r[7] = (bf16)b.w;
    *(v8bf*)(out + i) = r;
}

// ---------------------------------------------------------------------------
// Kernel 0b: V[b][j][h] f32 -> VT[b][h][jpad] bf16 (zero-padded to SPK)
// grid = (SPK/32, HIDDEN/32, BATCH), block = 256, LDS transpose
// ---------------------------------------------------------------------------
__global__ __launch_bounds__(256)
void vtrans_kernel(const float* __restrict__ v, bf16* __restrict__ vt, int S, int SPK)
{
    __shared__ float tile[32][33];
    const int j0 = blockIdx.x * 32;
    const int h0 = blockIdx.y * 32;
    const int b  = blockIdx.z;
    const int tl = threadIdx.x & 31;
    const int tg = threadIdx.x >> 5;   // 8 groups of 32 lanes

    #pragma unroll
    for (int r = 0; r < 4; ++r) {
        int jl = tg * 4 + r;
        int j  = j0 + jl;
        int h  = h0 + tl;
        tile[jl][tl] = (j < S) ? v[((size_t)b * S + j) * HIDDEN + h] : 0.f;
    }
    __syncthreads();
    #pragma unroll
    for (int r = 0; r < 4; ++r) {
        int hl = tg * 4 + r;
        int h  = h0 + hl;
        int j  = j0 + tl;
        vt[((size_t)b * HIDDEN + h) * SPK + j] = (bf16)tile[tl][hl];
    }
}

// ---------------------------------------------------------------------------
// Kernel 1: scores = Qb Kb^T * scale (bf16 WMMA), row softmax.
//   - writes attn (f32) to the output's attn region
//   - writes zero-padded bf16 attn [B][SP][SPK] to workspace for kernel 2
// grid = (numItiles, BATCH), block = 32 * numJtiles
// ---------------------------------------------------------------------------
__global__ __launch_bounds__(256)
void attn_softmax_kernel(const bf16* __restrict__ qb, const bf16* __restrict__ kb,
                         float* __restrict__ attn_out, bf16* __restrict__ attn_pad,
                         int S, int SP, int SPK, float scale)
{
    __shared__ float sc[16][136];
    __shared__ float rinv_s[16];

    const int itile = blockIdx.x;
    const int b     = blockIdx.y;
    const int w     = threadIdx.x >> 5;
    const int lane  = threadIdx.x & 31;
    const int half  = lane >> 4;
    const int l16   = lane & 15;
    const int i0    = itile * 16;
    const int j0    = w * 16;

    const int arow = min(i0 + l16, S - 1);
    const int brow = min(j0 + l16, S - 1);
    const bf16* qrow = qb + ((size_t)b * S + arow) * HIDDEN;
    const bf16* krow = kb + ((size_t)b * S + brow) * HIDDEN;
    const int aoff = half * 8;    // A fragment: two runs of 8
    const int boff = half * 8;    // B fragment: two runs of 8

    v8f acc = {};
    for (int hc = 0; hc < HIDDEN; hc += 32) {
        v16bf af = join16(*(const v8bf*)(qrow + hc + aoff),
                          *(const v8bf*)(qrow + hc + aoff + 16));
        v16bf bfr = join16(*(const v8bf*)(krow + hc + boff),
                           *(const v8bf*)(krow + hc + boff + 16));
        acc = __builtin_amdgcn_wmma_f32_16x16x32_bf16(false, af, false, bfr,
                                                      (short)0, acc, false, false);
    }

    // C layout: acc[r] -> row (r + 8*half), col l16 of this wave's j-tile
    #pragma unroll
    for (int r = 0; r < 8; ++r) {
        int li = r + half * 8;
        int cj = j0 + l16;
        sc[li][cj] = (cj < S) ? acc[r] * scale : -1e30f;
    }
    __syncthreads();

    if (threadIdx.x < 16) {
        int t = threadIdx.x;
        float m = -1e30f;
        for (int j = 0; j < S; ++j) m = fmaxf(m, sc[t][j]);
        float s = 0.f;
        for (int j = 0; j < S; ++j) {
            float e = __expf(sc[t][j] - m);
            sc[t][j] = e;
            s += e;
        }
        rinv_s[t] = 1.0f / s;
    }
    __syncthreads();

    for (int idx = threadIdx.x; idx < 16 * SPK; idx += blockDim.x) {
        int r = idx / SPK, c = idx % SPK;
        int ig = i0 + r;
        float a = (c < S) ? sc[r][c] * rinv_s[r] : 0.f;
        bf16 ab = (ig < S) ? (bf16)a : (bf16)0.f;
        attn_pad[((size_t)b * SP + ig) * SPK + c] = ab;
        if (ig < S && c < S)
            attn_out[((size_t)b * S + ig) * S + c] = a;
    }
}

// ---------------------------------------------------------------------------
// Kernel 2: S_all = attn @ V (bf16 WMMA using transposed VT), then the
// 6-region weighted combine in f32.
// grid = (numItiles, BATCH), block = 256 (8 waves x 4 h-tiles each)
// ---------------------------------------------------------------------------
__global__ __launch_bounds__(256)
void attn_apply_kernel(const float* __restrict__ v, const float* __restrict__ aw,
                       const float* __restrict__ attnF, const bf16* __restrict__ attnP,
                       const bf16* __restrict__ vt, float* __restrict__ out,
                       const int* __restrict__ Kptr, int S, int SP, int SPK)
{
    const int itile = blockIdx.x;
    const int b     = blockIdx.y;
    const int w     = threadIdx.x >> 5;
    const int lane  = threadIdx.x & 31;
    const int half  = lane >> 4;
    const int l16   = lane & 15;
    const int i0    = itile * 16;
    const int NK    = S - 1;
    const int Kc    = Kptr[0];

    v8f acc[4] = {v8f{}, v8f{}, v8f{}, v8f{}};
    const bf16* arow = attnP + ((size_t)b * SP + (i0 + l16)) * SPK;
    const int aoff = half * 8;
    const int boff = half * 8;

    for (int jc = 0; jc < SPK; jc += 32) {
        v16bf af = join16(*(const v8bf*)(arow + jc + aoff),
                          *(const v8bf*)(arow + jc + aoff + 16));
        #pragma unroll
        for (int t = 0; t < 4; ++t) {
            int hcol = (w * 4 + t) * 16 + l16;
            const bf16* vrow = vt + ((size_t)b * HIDDEN + hcol) * SPK;
            v16bf bfr = join16(*(const v8bf*)(vrow + jc + boff),
                               *(const v8bf*)(vrow + jc + boff + 16));
            acc[t] = __builtin_amdgcn_wmma_f32_16x16x32_bf16(false, af, false, bfr,
                                                             (short)0, acc[t], false, false);
        }
    }

    // Region-weighted combine:
    //  i<NK : w2*Sall + (w1-w2)*Sclass + (w0-w1)*attn[i,i]*v[i] + (w3-w2)*attn[i,NK]*v[NK]
    //  i=NK : w4*(Sall - aNN*v[NK]) + w5*aNN*v[NK]
    #pragma unroll
    for (int t = 0; t < 4; ++t) {
        int h = (w * 4 + t) * 16 + l16;
        float w0h = aw[0 * HIDDEN + h], w1h = aw[1 * HIDDEN + h];
        float w2h = aw[2 * HIDDEN + h], w3h = aw[3 * HIDDEN + h];
        float w4h = aw[4 * HIDDEN + h], w5h = aw[5 * HIDDEN + h];
        float vNK = v[((size_t)b * S + NK) * HIDDEN + h];
        #pragma unroll
        for (int r = 0; r < 8; ++r) {
            int ig = i0 + r + half * 8;
            if (ig >= S) continue;
            float sall = acc[t][r];
            const float* aF = attnF + ((size_t)b * S + ig) * S;
            float res;
            if (ig < NK) {
                float adiag = aF[ig];
                float aiNK  = aF[NK];
                float vii   = v[((size_t)b * S + ig) * HIDDEN + h];
                int c0 = (ig / Kc) * Kc;
                float scls = 0.f;
                for (int u = 0; u < Kc; ++u)
                    scls += aF[c0 + u] * v[((size_t)b * S + c0 + u) * HIDDEN + h];
                res = w2h * sall + (w1h - w2h) * scls
                    + (w0h - w1h) * adiag * vii + (w3h - w2h) * aiNK * vNK;
            } else {
                float aNN = aF[NK];
                res = w4h * (sall - aNN * vNK) + w5h * aNN * vNK;
            }
            out[((size_t)b * S + ig) * HIDDEN + h] = res;
        }
    }
}

extern "C" void kernel_launch(void* const* d_in, const int* in_sizes, int n_in,
                              void* d_out, int out_size, void* d_ws, size_t ws_size,
                              hipStream_t stream) {
    (void)n_in; (void)out_size; (void)ws_size;
    const float* q  = (const float*)d_in[0];
    const float* k  = (const float*)d_in[1];
    const float* v  = (const float*)d_in[2];
    const float* aw = (const float*)d_in[3];
    const int* Kptr = (const int*)d_in[5];

    const int S   = in_sizes[0] / (BATCH * HIDDEN);   // 101
    const int ITL = (S + 15) / 16;                    // 7 row tiles
    const int SP  = ITL * 16;                         // 112 padded rows
    const int SPK = ((S + 31) / 32) * 32;             // 128 padded cols (K-dim chunks)

    float* out      = (float*)d_out;
    float* attn_out = out + (size_t)BATCH * S * HIDDEN;

    // Workspace layout (bf16): attn_pad | Qb | Kb | VT
    const size_t nQK  = (size_t)BATCH * S * HIDDEN;
    bf16* attn_pad = (bf16*)d_ws;                              // B*SP*SPK
    bf16* qb       = attn_pad + (size_t)BATCH * SP * SPK;
    bf16* kb       = qb + nQK;
    bf16* vtw      = kb + nQK;                                 // B*HIDDEN*SPK
    const float scale = 1.0f / sqrtf((float)HIDDEN);

    const int cvtBlocks = (int)((nQK / 8 + 255) / 256);
    cvt_bf16_kernel<<<cvtBlocks, 256, 0, stream>>>(q, qb, nQK);
    cvt_bf16_kernel<<<cvtBlocks, 256, 0, stream>>>(k, kb, nQK);
    vtrans_kernel<<<dim3(SPK / 32, HIDDEN / 32, BATCH), 256, 0, stream>>>(v, vtw, S, SPK);

    dim3 grid(ITL, BATCH);
    attn_softmax_kernel<<<grid, 32 * ITL, 0, stream>>>(qb, kb, attn_out, attn_pad,
                                                       S, SP, SPK, scale);
    attn_apply_kernel<<<grid, 256, 0, stream>>>(v, aw, attn_out, attn_pad, vtw, out,
                                                Kptr, S, SP, SPK);
}
